// NeRFNetwork_68307159875808
// MI455X (gfx1250) — matrix operand order repaired and checked by reading, not compiled, not run on hardware
//
#include <hip/hip_runtime.h>

typedef _Float16 h16v __attribute__((ext_vector_type(16)));
typedef _Float16 h8v  __attribute__((ext_vector_type(8)));
typedef float    f8v  __attribute__((ext_vector_type(8)));

#define NLEV 16
#define TBL  524288
#define HASH_MASK (TBL - 1)
#define WPB 8           // waves per block
#define PTW 16          // points per wave (one WMMA M-tile)

__device__ __forceinline__ h16v cat16(h8v lo, h8v hi) {
  return __builtin_shufflevector(lo, hi, 0, 1, 2, 3, 4, 5, 6, 7,
                                 8, 9, 10, 11, 12, 13, 14, 15);
}

// A fragment (16x32 f16) for v_wmma_f32_16x16x32_f16 from an LDS activation
// buffer laid out [16 points][rowHalves feats], at feature offset kBase.
// ISA layout: lanes 0-15 -> M=lane, K = {0..7,16..23}; lanes 16-31 -> K = {8..15,24..31}.
__device__ __forceinline__ h16v ldsA(const _Float16* buf, int rowHalves, int kBase, int lane) {
  int o = (lane & 15) * rowHalves + kBase + ((lane & 16) ? 8 : 0);
  h8v lo = *(const h8v*)(buf + o);
  h8v hi = *(const h8v*)(buf + o + 16);
  return cat16(lo, hi);
}

// B fragment (32x16 f16): weights stored row-major [outRow][rowHalves feats].
// lane -> column N = colBase + lane%16; K = kBase + (lane<16 ? {0..7,16..23} : {8..15,24..31}).
__device__ __forceinline__ h16v ldsB(const _Float16* w, int rowHalves, int colBase, int kBase, int lane) {
  int o = (colBase + (lane & 15)) * rowHalves + kBase + ((lane & 16) ? 8 : 0);
  h8v lo = *(const h8v*)(w + o);
  h8v hi = *(const h8v*)(w + o + 16);
  return cat16(lo, hi);
}

__device__ __forceinline__ f8v wmma_f16(h16v a, h16v b, f8v c) {
  return __builtin_amdgcn_wmma_f32_16x16x32_f16(false, a, false, b, (short)0, c,
                                                false, false);
}

// sum across the two half-waves (lane <-> lane^16) via ds_swizzle SWAPX16
// (group-of-32 encoding: xor_mask=0x10 -> offset 0x401F). No LDS memory touched.
__device__ __forceinline__ float xhalf_sum(float v) {
  int sw = __builtin_amdgcn_ds_swizzle(__builtin_bit_cast(int, v), 0x401F);
  return v + __builtin_bit_cast(float, sw);
}

__device__ __forceinline__ unsigned pack_h2(float a, float b) {
  unsigned lo = (unsigned)__builtin_bit_cast(unsigned short, (_Float16)a);
  unsigned hi = (unsigned)__builtin_bit_cast(unsigned short, (_Float16)b);
  return lo | (hi << 16);
}

__global__ void __launch_bounds__(256)
nerf_fused(const float* __restrict__ gx, const float* __restrict__ gd,
           const float* __restrict__ gtab,
           const float* __restrict__ gw1, const float* __restrict__ gw2,
           const float* __restrict__ gc1, const float* __restrict__ gc2,
           const float* __restrict__ gc3,
           float* __restrict__ gout, int npts) {
  __shared__ __align__(16) _Float16 sW1[64 * 32];  // w_sig1 [64][32]
  __shared__ __align__(16) _Float16 sW2[16 * 64];  // w_sig2 [16][64]
  __shared__ __align__(16) _Float16 sC1[64 * 32];  // w_col1 [64][31] padded K->32
  __shared__ __align__(16) _Float16 sC2[64 * 64];  // w_col2 [64][64]
  __shared__ __align__(16) _Float16 sC3[16 * 64];  // w_col3 [3][64] padded rows->16
  __shared__ __align__(16) _Float16 sStg[WPB][PTW * 64];  // per-wave staging

  const int tid = threadIdx.x;
  // --- stage weights as f16 into LDS (once per block) ---
  for (int i = tid; i < 64 * 32; i += 256) sW1[i] = (_Float16)gw1[i];
  for (int i = tid; i < 16 * 64; i += 256) sW2[i] = (_Float16)gw2[i];
  for (int i = tid; i < 64 * 32; i += 256) {
    int r = i >> 5, k = i & 31;
    sC1[i] = (k < 31) ? (_Float16)gc1[r * 31 + k] : (_Float16)0.0f;
  }
  for (int i = tid; i < 64 * 64; i += 256) sC2[i] = (_Float16)gc2[i];
  for (int i = tid; i < 16 * 64; i += 256) {
    int r = i >> 6, k = i & 63;
    sC3[i] = (r < 3) ? (_Float16)gc3[r * 64 + k] : (_Float16)0.0f;
  }
  __syncthreads();

  const int lane = tid & 31;
  const int wv = tid >> 5;
  const int n16 = lane & 15;               // point / output-column selector
  const int hiH = (lane >> 4) & 1;         // half-wave id (= corner x-bit)
  const int ptBase = (blockIdx.x * WPB + wv) * PTW;
  if (ptBase + PTW > npts) return;         // wave-uniform
  _Float16* st = sStg[wv];

  // ================= hash-grid encoding =================
  // Wave-uniform over levels (compile-time res, dense/hash folded). Lane owns
  // point n16; each half-wave gathers 4 of the 8 trilinear corners (x-offset =
  // hiH); half-wave partials combined with ds_swizzle SWAPX16. The 64 MB table
  // is L2-resident on MI455X, so these b64 gathers are L2-latency bound —
  // straight-line unrolled code maximizes loads in flight.
  {
    static constexpr int RESC[NLEV] = {16, 22, 30, 42, 58, 80, 111, 153,
                                       212, 294, 406, 561, 776, 1072, 1482, 2048};
    const int pt = ptBase + n16;
    const float u0 = (gx[pt * 3 + 0] + 1.0f) * 0.5f;
    const float u1 = (gx[pt * 3 + 1] + 1.0f) * 0.5f;
    const float u2 = (gx[pt * 3 + 2] + 1.0f) * 0.5f;
    unsigned pk[NLEV];
#pragma unroll
    for (int l = 0; l < NLEV; ++l) {
      const int res = RESC[l];
      const int rp1 = res + 1;
      const bool dense = ((long long)rp1 * rp1 * rp1) <= (long long)TBL;
      const float fres = (float)res;
      float px = u0 * fres, py = u1 * fres, pz = u2 * fres;
      float w0x = floorf(px), w0y = floorf(py), w0z = floorf(pz);
      int ix = (int)w0x, iy = (int)w0y, iz = (int)w0z;
      float fx = px - w0x, fy = py - w0y, fz = pz - w0z;
      const float wx = hiH ? fx : 1.0f - fx;     // this half-wave's x-weight
      const unsigned cx = (unsigned)(ix + hiH);
      const float* tl = gtab + (size_t)l * TBL * 2;
      float a0 = 0.0f, a1 = 0.0f;
#pragma unroll
      for (int c = 0; c < 4; ++c) {              // 4 yz-corners per half-wave
        int oy = (c >> 1) & 1, oz = c & 1;
        unsigned cy = (unsigned)(iy + oy);
        unsigned cz = (unsigned)(iz + oz);
        unsigned idx = dense
            ? (cx + (unsigned)rp1 * (cy + (unsigned)rp1 * cz))
            : ((cx ^ (cy * 2654435761u) ^ (cz * 805459861u)) & HASH_MASK);
        float w = wx * (oy ? fy : 1.0f - fy) * (oz ? fz : 1.0f - fz);
        const float* fp = tl + (size_t)idx * 2;
        a0 += w * fp[0];
        a1 += w * fp[1];
      }
      a0 = xhalf_sum(a0);                        // combine the two half-waves
      a1 = xhalf_sum(a1);
      pk[l] = pack_h2(a0, a1);                   // f16 feature pair of level l
    }
    // lanes 0-15 commit the full 32-feature f16 row of their point: 4x b128
    if (lane < 16) {
      uint4* dst = (uint4*)(st + n16 * 32);
      dst[0] = make_uint4(pk[0], pk[1], pk[2], pk[3]);
      dst[1] = make_uint4(pk[4], pk[5], pk[6], pk[7]);
      dst[2] = make_uint4(pk[8], pk[9], pk[10], pk[11]);
      dst[3] = make_uint4(pk[12], pk[13], pk[14], pk[15]);
    }
  }

  const f8v z8 = {0.f, 0.f, 0.f, 0.f, 0.f, 0.f, 0.f, 0.f};

  // ================= sigma MLP layer 1: enc[16x32] @ w_sig1^T -> h1[16x64] =================
  f8v h1[4];
  {
    h16v a = ldsA(st, 32, 0, lane);
#pragma unroll
    for (int nb = 0; nb < 4; ++nb)
      h1[nb] = wmma_f16(a, ldsB(sW1, 32, nb * 16, 0, lane), z8);
  }
  // ReLU + restage as [pt][64] f16 (C layout: lane=col n, vgpr r -> row r+8*hiH)
#pragma unroll
  for (int nb = 0; nb < 4; ++nb) {
#pragma unroll
    for (int r = 0; r < 8; ++r) {
      float v = fmaxf(h1[nb][r], 0.0f);
      st[(r + 8 * hiH) * 64 + nb * 16 + n16] = (_Float16)v;
    }
  }

  // ================= sigma MLP layer 2: h1[16x64] @ w_sig2^T -> h2[16x16] =================
  f8v h2 = z8;
  {
    h16v a0 = ldsA(st, 64, 0, lane);
    h16v a1 = ldsA(st, 64, 32, lane);
    h2 = wmma_f16(a0, ldsB(sW2, 64, 0, 0, lane), h2);
    h2 = wmma_f16(a1, ldsB(sW2, 64, 0, 32, lane), h2);
  }

  // sigma = exp(h2[:,0]) — column-0 lanes write it out
  if (n16 == 0) {
#pragma unroll
    for (int r = 0; r < 8; ++r) {
      int pt = ptBase + r + 8 * hiH;
      gout[pt] = expf(h2[r]);
    }
  }

  // ================= build hc = [SH(16) | geo(15) | pad] as [pt][32] f16 =================
  if (lane < 16) {
    int pt = ptBase + lane;
    float x = gd[pt * 3 + 0], y = gd[pt * 3 + 1], z = gd[pt * 3 + 2];
    float xx = x * x, yy = y * y, zz = z * z;
    float xy = x * y, yz = y * z, xz = x * z;
    float s[16];
    s[0] = 0.28209479177387814f;
    s[1] = -0.48860251190291987f * y;
    s[2] = 0.48860251190291987f * z;
    s[3] = -0.48860251190291987f * x;
    s[4] = 1.0925484305920792f * xy;
    s[5] = -1.0925484305920792f * yz;
    s[6] = 0.94617469575756f * zz - 0.31539156525252f;
    s[7] = -1.0925484305920792f * xz;
    s[8] = 0.5462742152960396f * xx - 0.5462742152960396f * yy;
    s[9] = 0.5900435899266435f * y * (-3.0f * xx + yy);
    s[10] = 2.890611442640554f * xy * z;
    s[11] = 0.4570457994644657f * y * (1.0f - 5.0f * zz);
    s[12] = 0.3731763325901154f * z * (5.0f * zz - 3.0f);
    s[13] = 0.4570457994644657f * x * (1.0f - 5.0f * zz);
    s[14] = 1.445305721320277f * z * (xx - yy);
    s[15] = 0.5900435899266435f * x * (-xx + 3.0f * yy);
    unsigned sp[8];
#pragma unroll
    for (int i = 0; i < 8; ++i) sp[i] = pack_h2(s[2 * i], s[2 * i + 1]);
    uint4* dst = (uint4*)(st + lane * 32);
    dst[0] = make_uint4(sp[0], sp[1], sp[2], sp[3]);
    dst[1] = make_uint4(sp[4], sp[5], sp[6], sp[7]);
  }
  if (n16 > 0) {  // geo = h2[:,1:16] -> cols 16..30
#pragma unroll
    for (int r = 0; r < 8; ++r)
      st[(r + 8 * hiH) * 32 + 16 + (n16 - 1)] = (_Float16)h2[r];
  } else {        // pad col 31 with zeros
#pragma unroll
    for (int r = 0; r < 8; ++r)
      st[(r + 8 * hiH) * 32 + 31] = (_Float16)0.0f;
  }

  // ================= color layer 1: hc[16x32] @ w_col1^T -> [16x64], ReLU ===========
  f8v hc1[4];
  {
    h16v a = ldsA(st, 32, 0, lane);
#pragma unroll
    for (int nb = 0; nb < 4; ++nb)
      hc1[nb] = wmma_f16(a, ldsB(sC1, 32, nb * 16, 0, lane), z8);
  }
#pragma unroll
  for (int nb = 0; nb < 4; ++nb) {
#pragma unroll
    for (int r = 0; r < 8; ++r) {
      float v = fmaxf(hc1[nb][r], 0.0f);
      st[(r + 8 * hiH) * 64 + nb * 16 + n16] = (_Float16)v;
    }
  }

  // ================= color layer 2: [16x64] @ w_col2^T -> [16x64], ReLU =============
  f8v hc2[4];
  {
    h16v a0 = ldsA(st, 64, 0, lane);
    h16v a1 = ldsA(st, 64, 32, lane);
#pragma unroll
    for (int nb = 0; nb < 4; ++nb) {
      f8v acc = wmma_f16(a0, ldsB(sC2, 64, nb * 16, 0, lane), z8);
      hc2[nb] = wmma_f16(a1, ldsB(sC2, 64, nb * 16, 32, lane), acc);
    }
  }
#pragma unroll
  for (int nb = 0; nb < 4; ++nb) {
#pragma unroll
    for (int r = 0; r < 8; ++r) {
      float v = fmaxf(hc2[nb][r], 0.0f);
      st[(r + 8 * hiH) * 64 + nb * 16 + n16] = (_Float16)v;
    }
  }

  // ================= color layer 3: [16x64] @ w_col3^T(pad 16) -> [16x16] ===========
  f8v o = z8;
  {
    h16v a0 = ldsA(st, 64, 0, lane);
    h16v a1 = ldsA(st, 64, 32, lane);
    o = wmma_f16(a0, ldsB(sC3, 64, 0, 0, lane), o);
    o = wmma_f16(a1, ldsB(sC3, 64, 0, 32, lane), o);
  }
  if (n16 < 3) {  // only columns 0..2 are real outputs
#pragma unroll
    for (int r = 0; r < 8; ++r) {
      int pt = ptBase + r + 8 * hiH;
      float v = 1.0f / (1.0f + expf(-o[r]));
      gout[npts + pt * 3 + n16] = v;  // color flat after sigma block
    }
  }
}

extern "C" void kernel_launch(void* const* d_in, const int* in_sizes, int n_in,
                              void* d_out, int out_size, void* d_ws, size_t ws_size,
                              hipStream_t stream) {
  (void)n_in; (void)out_size; (void)d_ws; (void)ws_size;
  const float* x = (const float*)d_in[0];
  const float* d = (const float*)d_in[1];
  const float* table = (const float*)d_in[2];
  const float* w_sig1 = (const float*)d_in[3];
  const float* w_sig2 = (const float*)d_in[4];
  const float* w_col1 = (const float*)d_in[5];
  const float* w_col2 = (const float*)d_in[6];
  const float* w_col3 = (const float*)d_in[7];
  float* out = (float*)d_out;

  int npts = in_sizes[0] / 3;                     // 1048576
  int waves = (npts + PTW - 1) / PTW;             // 16 points per wave
  int blocks = (waves + WPB - 1) / WPB;           // 8 waves per 256-thread block
  nerf_fused<<<blocks, 256, 0, stream>>>(x, d, table, w_sig1, w_sig2,
                                         w_col1, w_col2, w_col3, out, npts);
}
